// MultiHeadAttention_9826885173883
// MI455X (gfx1250) — matrix operand order repaired and checked by reading, not compiled, not run on hardware
//
#include <hip/hip_runtime.h>

#define NUM_HEADS 16
#define HEAD_DIM  64
#define BSZ       2
#define SEQ       2048
#define EMBED     (NUM_HEADS * HEAD_DIM)
#define SCALE     0.125f                 // 64^-0.5
#define LOG2E     1.4426950408889634f

typedef __attribute__((ext_vector_type(16))) _Float16 v16h;
typedef __attribute__((ext_vector_type(8)))  _Float16 v8h;
typedef __attribute__((ext_vector_type(2)))  _Float16 v2h;
typedef __attribute__((ext_vector_type(8)))  float    v8f;
typedef __attribute__((ext_vector_type(4)))  float    v4f;

// wave32 XOR lane shuffle via ds_swizzle (group-of-32 mode:
// offset = {xor[4:0]<<10 | or<<5 | and[4:0]}, and=0x1F keeps lane id).
template <int XM>
__device__ __forceinline__ float swz(float x) {
  return __int_as_float(
      __builtin_amdgcn_ds_swizzle(__float_as_int(x), (XM << 10) | 0x1F));
}

__device__ __forceinline__ v8f vzero8() {
  v8f z;
#pragma unroll
  for (int e = 0; e < 8; ++e) z[e] = 0.0f;
  return z;
}

// max/sum across the 16 lanes of each half-wave (xor masks 1,2,4,8 stay
// inside lanes [0,15] and [16,31] — exactly the C/D-layout row groups).
__device__ __forceinline__ v8f red16_max(v8f x) {
#pragma unroll
  for (int e = 0; e < 8; ++e) x[e] = fmaxf(x[e], swz<1>(x[e]));
#pragma unroll
  for (int e = 0; e < 8; ++e) x[e] = fmaxf(x[e], swz<2>(x[e]));
#pragma unroll
  for (int e = 0; e < 8; ++e) x[e] = fmaxf(x[e], swz<4>(x[e]));
#pragma unroll
  for (int e = 0; e < 8; ++e) x[e] = fmaxf(x[e], swz<8>(x[e]));
  return x;
}

__device__ __forceinline__ v8f red16_sum(v8f x) {
#pragma unroll
  for (int e = 0; e < 8; ++e) x[e] += swz<1>(x[e]);
#pragma unroll
  for (int e = 0; e < 8; ++e) x[e] += swz<2>(x[e]);
#pragma unroll
  for (int e = 0; e < 8; ++e) x[e] += swz<4>(x[e]);
#pragma unroll
  for (int e = 0; e < 8; ++e) x[e] += swz<8>(x[e]);
  return x;
}

__global__ __launch_bounds__(128) void fa_fwd_f16wmma(
    const float* __restrict__ Qg, const float* __restrict__ Kg,
    const float* __restrict__ Vg, float* __restrict__ Og) {
  // K tile: [key][d] f16, V tile transposed: [d][key] f16, P scratch per wave.
  __shared__ __align__(32) _Float16 sK[64 * 64];
  __shared__ __align__(32) _Float16 sVt[64 * 64];
  __shared__ __align__(32) _Float16 sP[4 * 16 * 64];

  const int b = blockIdx.z, h = blockIdx.y, qb = blockIdx.x;
  const int tid = threadIdx.x;
  const int wave = tid >> 5, lane = tid & 31;
  const int lane16 = lane & 15, hi = lane >> 4;
  const int q0 = qb * 64 + wave * 16;  // this wave's 16 query rows
  const int nk = qb + 1;               // causal: k-tiles 0..qb

  // ---- Load Q A-fragments once (contiguous k order: half j -> k = hi*16 + j)
  const float* qrow =
      Qg + (size_t)(b * SEQ + q0 + lane16) * EMBED + h * HEAD_DIM;
  v16h qf[2];
#pragma unroll
  for (int c = 0; c < 2; ++c) {
    const int d0 = c * 32 + hi * 16;
    v4f a0 = *(const v4f*)(qrow + d0);
    v4f a1 = *(const v4f*)(qrow + d0 + 4);
    v4f a2 = *(const v4f*)(qrow + d0 + 8);
    v4f a3 = *(const v4f*)(qrow + d0 + 12);
#pragma unroll
    for (int j = 0; j < 4; ++j) {
      qf[c][j]      = (_Float16)(a0[j] * SCALE);
      qf[c][j + 4]  = (_Float16)(a1[j] * SCALE);
      qf[c][j + 8]  = (_Float16)(a2[j] * SCALE);
      qf[c][j + 12] = (_Float16)(a3[j] * SCALE);
    }
  }

  // ---- Flash-attention state (C/D layout: vgpr e ↔ row e + hi*8)
  v8f m_run, l_run, Oacc[4];
#pragma unroll
  for (int e = 0; e < 8; ++e) { m_run[e] = -3.0e38f; l_run[e] = 0.0f; }
#pragma unroll
  for (int nt = 0; nt < 4; ++nt) Oacc[nt] = vzero8();

  for (int kt = 0; kt < nk; ++kt) {
    __syncthreads();  // previous tile's LDS consumers done before restage
    const int k0 = kt * 64;

    {  // stage K tile: 2 threads per key row, f32->f16, row-major
      const int kr = tid >> 1, dp = (tid & 1) * 32;
      const float* src =
          Kg + (size_t)(b * SEQ + k0 + kr) * EMBED + h * HEAD_DIM + dp;
      _Float16* dst = &sK[kr * 64 + dp];
#pragma unroll
      for (int g = 0; g < 4; ++g) {
        v4f f0 = *(const v4f*)(src + g * 8);
        v4f f1 = *(const v4f*)(src + g * 8 + 4);
        v8h hv;
#pragma unroll
        for (int j = 0; j < 4; ++j) {
          hv[j] = (_Float16)f0[j];
          hv[j + 4] = (_Float16)f1[j];
        }
        *(v8h*)(dst + g * 8) = hv;
      }
      if (kt + 1 < nk) __builtin_prefetch(src + (size_t)64 * EMBED, 0, 1);
    }
    {  // stage V tile transposed -> sVt[d][key]; pack 2 keys per b32 store
      const int kp = (tid & 31) * 2, dg = (tid >> 5) * 16;
      const float* sA =
          Vg + (size_t)(b * SEQ + k0 + kp) * EMBED + h * HEAD_DIM + dg;
      const float* sB = sA + EMBED;
      v4f fa[4], fb[4];
#pragma unroll
      for (int g = 0; g < 4; ++g) {
        fa[g] = *(const v4f*)(sA + g * 4);
        fb[g] = *(const v4f*)(sB + g * 4);
      }
#pragma unroll
      for (int d = 0; d < 16; ++d) {
        v2h pr;
        pr[0] = (_Float16)fa[d >> 2][d & 3];
        pr[1] = (_Float16)fb[d >> 2][d & 3];
        *(v2h*)&sVt[(dg + d) * 64 + kp] = pr;
      }
      if (kt + 1 < nk) __builtin_prefetch(sA + (size_t)64 * EMBED, 0, 1);
    }
    __syncthreads();

    // ---- GEMM1: S(16x64) = Q(16x64) * K^T ; B-frag = 16 contiguous halves
    v8f s[4];
#pragma unroll
    for (int nt = 0; nt < 4; ++nt) {
      v8f acc = vzero8();
#pragma unroll
      for (int c = 0; c < 2; ++c) {
        v16h bf = *(const v16h*)&sK[(nt * 16 + lane16) * 64 + c * 32 + hi * 16];
        acc = __builtin_amdgcn_wmma_f32_16x16x32_f16(
            false, qf[c], false, bf, (short)0, acc, false, false);
      }
      s[nt] = acc;
    }

    // ---- causal mask (only diagonal tile kt == qb can be masked)
    if (kt == qb) {
#pragma unroll
      for (int nt = 0; nt < 4; ++nt)
#pragma unroll
        for (int e = 0; e < 8; ++e) {
          const int key = k0 + nt * 16 + lane16;
          const int row = q0 + e + hi * 8;
          if (key > row) s[nt][e] = -1.0e9f;
        }
    }

    // ---- online softmax
    v8f rmax = s[0];
#pragma unroll
    for (int nt = 1; nt < 4; ++nt)
#pragma unroll
      for (int e = 0; e < 8; ++e) rmax[e] = fmaxf(rmax[e], s[nt][e]);
    rmax = red16_max(rmax);

    v8f m_new, alpha;
#pragma unroll
    for (int e = 0; e < 8; ++e) {
      m_new[e] = fmaxf(m_run[e], rmax[e]);
      alpha[e] = __builtin_amdgcn_exp2f((m_run[e] - m_new[e]) * LOG2E);
    }
    v8f p[4];
#pragma unroll
    for (int nt = 0; nt < 4; ++nt)
#pragma unroll
      for (int e = 0; e < 8; ++e)
        p[nt][e] = __builtin_amdgcn_exp2f((s[nt][e] - m_new[e]) * LOG2E);

    v8f rsum;
#pragma unroll
    for (int e = 0; e < 8; ++e)
      rsum[e] = p[0][e] + p[1][e] + p[2][e] + p[3][e];
    rsum = red16_sum(rsum);

#pragma unroll
    for (int e = 0; e < 8; ++e) {
      l_run[e] = l_run[e] * alpha[e] + rsum[e];
      m_run[e] = m_new[e];
    }
#pragma unroll
    for (int nt = 0; nt < 4; ++nt)
#pragma unroll
      for (int e = 0; e < 8; ++e) Oacc[nt][e] *= alpha[e];

    // ---- P: C/D layout -> f16 LDS -> A layout
    _Float16* pw = &sP[wave * 16 * 64];
#pragma unroll
    for (int nt = 0; nt < 4; ++nt)
#pragma unroll
      for (int e = 0; e < 8; ++e)
        pw[(e + hi * 8) * 64 + nt * 16 + lane16] = (_Float16)p[nt][e];
    __syncthreads();

    v16h pf[2];
#pragma unroll
    for (int c = 0; c < 2; ++c) {
      v8h lo = *(const v8h*)&pw[lane16 * 64 + c * 32 + hi * 16];
      v8h hh = *(const v8h*)&pw[lane16 * 64 + c * 32 + hi * 16 + 8];
#pragma unroll
      for (int j = 0; j < 8; ++j) {
        pf[c][j] = lo[j];
        pf[c][j + 8] = hh[j];
      }
    }

    // ---- GEMM2: O(16x64) += P(16x64) * V(64x64); B-frag from sVt[d][key]
#pragma unroll
    for (int nt = 0; nt < 4; ++nt)
#pragma unroll
      for (int c = 0; c < 2; ++c) {
        v16h vb =
            *(const v16h*)&sVt[(nt * 16 + lane16) * 64 + c * 32 + hi * 16];
        Oacc[nt] = __builtin_amdgcn_wmma_f32_16x16x32_f16(
            false, pf[c], false, vb, (short)0, Oacc[nt], false, false);
      }
  }

  // ---- epilogue: normalize and store fp32
  v8f inv;
#pragma unroll
  for (int e = 0; e < 8; ++e) inv[e] = __builtin_amdgcn_rcpf(l_run[e]);

  float* orow = Og + (size_t)(b * SEQ + q0) * EMBED + h * HEAD_DIM;
#pragma unroll
  for (int nt = 0; nt < 4; ++nt)
#pragma unroll
    for (int e = 0; e < 8; ++e)
      orow[(size_t)(e + hi * 8) * EMBED + nt * 16 + lane16] =
          Oacc[nt][e] * inv[e];
}

extern "C" void kernel_launch(void* const* d_in, const int* in_sizes, int n_in,
                              void* d_out, int out_size, void* d_ws,
                              size_t ws_size, hipStream_t stream) {
  (void)in_sizes; (void)n_in; (void)out_size; (void)d_ws; (void)ws_size;
  const float* q = (const float*)d_in[0];
  const float* k = (const float*)d_in[1];
  const float* v = (const float*)d_in[2];
  // d_in[3] (additive causal mask) is implemented analytically in-kernel.
  float* out = (float*)d_out;
  dim3 grid(SEQ / 64, NUM_HEADS, BSZ);
  fa_fwd_f16wmma<<<grid, 128, 0, stream>>>(q, k, v, out);
}
